// Maron_3058016715313
// MI455X (gfx1250) — matrix operand order repaired and checked by reading, not compiled, not run on hardware
//
#include <hip/hip_runtime.h>
#include <stdint.h>

typedef __attribute__((ext_vector_type(16))) _Float16 v16h;
typedef __attribute__((ext_vector_type(8)))  float    v8f;

// Exponent table: all (e0,e1,e2,e3) >= 0 with sum 7, lexicographic order
// (matches sorted() of the reference's permutation-of-partition set; the four
// exponents land in feature columns {0,1,7,8}). Packed 8 bits per exponent.
struct FTab { uint32_t v[128]; };
static constexpr FTab make_ftab() {
    FTab t{};
    int idx = 0;
    for (int a = 0; a <= 7; ++a)
        for (int b = 0; b <= 7 - a; ++b)
            for (int c = 0; c <= 7 - a - b; ++c) {
                int d = 7 - a - b - c;
                t.v[idx++] = (uint32_t)a | ((uint32_t)b << 8) |
                             ((uint32_t)c << 16) | ((uint32_t)d << 24);
            }
    while (idx < 128) t.v[idx++] = 0u;
    return t;
}
__constant__ FTab c_f4 = make_ftab();

// ---- hardware tanh (V_TANH_F32, TRANS op: co-executes with WMMA) ----
__device__ __forceinline__ float fast_tanh(float x) {
#if __has_builtin(__builtin_amdgcn_tanhf)
    return __builtin_amdgcn_tanhf(x);
#else
    return tanhf(x);
#endif
}

// ---- constant-index cross-lane read (always executed convergently) ----
__device__ __forceinline__ float rl(float v, int l) {
    return __int_as_float(__builtin_amdgcn_readlane(__float_as_int(v), l));
}

// ---- xor-butterfly add within 8-lane groups via DPP8 (pure VALU, no LDS) ----
template <int X>
__device__ __forceinline__ float dpp8_xor_add(float v) {
#if __has_builtin(__builtin_amdgcn_mov_dpp8)
    constexpr unsigned imm =
        ((0u ^ X) << 0)  | ((1u ^ X) << 3)  | ((2u ^ X) << 6)  | ((3u ^ X) << 9) |
        ((4u ^ X) << 12) | ((5u ^ X) << 15) | ((6u ^ X) << 18) | ((7u ^ X) << 21);
    int m = __builtin_amdgcn_mov_dpp8(__float_as_int(v), imm);
    return v + __int_as_float(m);
#else
    return v + __shfl_xor(v, X, 32);
#endif
}
__device__ __forceinline__ float grp8_sum(float v) {
    v = dpp8_xor_add<1>(v);
    v = dpp8_xor_add<2>(v);
    v = dpp8_xor_add<4>(v);
    return v;
}

__global__ __launch_bounds__(256)
void maron_inv_kernel(const float* __restrict__ x,
                      const float* __restrict__ mW1, const float* __restrict__ mb1,
                      const float* __restrict__ mW2, const float* __restrict__ mb2,
                      const float* __restrict__ fW1, const float* __restrict__ fb1,
                      const float* __restrict__ fW2, const float* __restrict__ fb2,
                      float* __restrict__ out, int rows, int T)
{
    __shared__ float s_pw[8][14 * 8];   // per-wave power tables  x_c^e
    __shared__ float s_q [8][120];      // per-wave q accumulator (row m of block)

    const int lane = threadIdx.x & 31;
    const int wib  = threadIdx.x >> 5;
    const int row  = blockIdx.x * 8 + wib;
    const int hi   = lane >> 4;         // half-wave select (K/M split)
    const int n    = lane & 15;         // N index / row-within-tile

    float* Pw = s_pw[wib];
    float* q  = s_q[wib];

    // ---- powers of this row's 14 features: Pw[c*8+e] = x_c^e ----
    if (lane < 14) {
        int b = row / T, t1 = row % T;          // x is (B,14,T)
        float xv = x[(b * 14 + lane) * T + t1];
        float p = 1.0f;
        #pragma unroll
        for (int e = 0; e < 8; ++e) { Pw[lane * 8 + e] = p; p *= xv; }
    }

    // ---- B fragment: mul_W1 (14x16) zero-padded to 32x16, f16 ----
    v16h bf;
    #pragma unroll
    for (int j = 0; j < 16; ++j) {
        int k = (j < 8) ? (j + hi * 8) : (16 + (j - 8) + hi * 8);
        bf[j] = (k < 14) ? (_Float16)mW1[k * 16 + n] : (_Float16)0.0f;
    }
    // ---- A base fragment: 1.0 for real columns (exponent 0), 0 for pad ----
    v16h abase;
    #pragma unroll
    for (int j = 0; j < 16; ++j) {
        int c = (j < 8) ? (j + hi * 8) : (16 + (j - 8) + hi * 8);
        abase[j] = (c < 14) ? (_Float16)1.0f : (_Float16)0.0f;
    }

    const float b1n = mb1[n];
    const float w2n = mW2[n];
    const float b2s = mb2[0];
    const _Float16 one_h = (_Float16)1.0f;

    __syncthreads();   // powers visible

    float lossAcc = 0.0f;

    for (int t = 0; t < 8; ++t) {
        int k = t * 16 + n; if (k > 119) k = 119;    // pad rows: harmless dup
        const uint32_t ee = c_f4.v[k];               // s-invariant: load once
        const int i0 = (int)(ee & 7),        i1 = (int)((ee >> 8) & 7);
        const int i2 = (int)((ee >> 16) & 7), i3 = (int)((ee >> 24) & 7);

        // lanes 0/16 accumulate kk = t*16 + hi*8 + i; invalid only t=7,hi=1
        const float vmask = ((n == 0) && (hi == 0 || t < 7)) ? 1.0f : 0.0f;

        float qacc[8] = {0, 0, 0, 0, 0, 0, 0, 0};

        for (int s = 0; s < 7; ++s) {
            const int col0 = (7 - s) % 7;
            const int col1 = (8 - s) % 7;
            const int col2 = 7 + (7 - s) % 7;
            const int col3 = 7 + (8 - s) % 7;

            // all lanes load all four powers of their own row k
            float p0 = Pw[col0 * 8 + i0];
            float p1 = Pw[col1 * 8 + i1];
            float p2 = Pw[col2 * 8 + i2];
            float p3 = Pw[col3 * 8 + i3];
            float mv = (p0 * p1) * (p2 * p3);        // mul[k] at lane n

            v16h a = abase;                          // branch-free A build
            a[0] = (_Float16)(hi ? p3 : p0);
            a[1] = hi ? one_h : (_Float16)p1;
            a[7] = hi ? one_h : (_Float16)p2;

            v8f c = {};
            v8f d = __builtin_amdgcn_wmma_f32_16x16x32_f16(
                        false, a, false, bf, (short)0, c, false, false);

            #pragma unroll
            for (int i = 0; i < 8; ++i) {
                float v = fast_tanh(d[i] + b1n) * w2n;
                float g = grp8_sum(v);
                float r8  = rl(g, 8);                // unconditional readlanes
                float r24 = rl(g, 24);
                float red = g + (hi ? r24 : r8);     // valid at lanes 0,16
                float mulnn = red + b2s;
                qacc[i] += mulnn;

                float mva = rl(mv, i);               // mul[t*16+i]
                float mvb = rl(mv, i + 8);           // mul[t*16+8+i]
                float mulv = hi ? mvb : mva;
                lossAcc = fmaf(vmask, fabsf(mulv - mulnn), lossAcc);
            }
        }
        // lane 0 owns q[t*16+0..7], lane 16 owns q[t*16+8..15]: 2x b128 store
        if (n == 0 && (hi == 0 || t < 7)) {
            float4* dst = (float4*)&q[t * 16 + hi * 8];
            dst[0] = make_float4(qacc[0], qacc[1], qacc[2], qacc[3]);
            dst[1] = make_float4(qacc[4], qacc[5], qacc[6], qacc[7]);
        }
    }

    __syncthreads();   // all q complete

    // ---- per-row loss output (lanes 0 and 16 hold the two halves) ----
    float Lr = rl(lossAcc, 0) + rl(lossAcc, 16);
    if (lane == 0) out[rows + row] = Lr * (1.0f / 120.0f);

    // ---- final feature MLP for the block's 8 rows, done by wave 0 as WMMA:
    //      acc[m,n] = sum_k q_m[k] * fW1[k,n],  M = 8 rows (pad 16), K = 120->128
    if (wib == 0) {
        v8f cacc = {};
        #pragma unroll
        for (int cc = 0; cc < 4; ++cc) {
            v16h af, bf2;
            #pragma unroll
            for (int j = 0; j < 16; ++j) {
                int kx = cc * 32 + ((j < 8) ? (j + hi * 8)
                                            : (16 + (j - 8) + hi * 8));
                int kxc = kx < 120 ? kx : 119;
                float av = s_q[n & 7][kxc];          // row m = n (valid m<8)
                af[j] = (n < 8 && kx < 120) ? (_Float16)av : (_Float16)0.0f;

                int kb  = cc * 32 + ((j < 8) ? (j + hi * 8) : (16 + (j - 8) + hi * 8));     // B: K rows of fW1
                int kbc = kb < 120 ? kb : 119;
                float bv = fW1[kbc * 16 + n];
                bf2[j] = (kb < 120) ? (_Float16)bv : (_Float16)0.0f;
            }
            cacc = __builtin_amdgcn_wmma_f32_16x16x32_f16(
                       false, af, false, bf2, (short)0, cacc, false, false);
        }
        const float fb1n = fb1[n];
        const float w2f  = fW2[n];
        const float fb2s = fb2[0];
        #pragma unroll
        for (int i = 0; i < 8; ++i) {                // rows m = i (hi=0 half)
            float hv = fast_tanh(cacc[i] + fb1n) * w2f;
            float g  = grp8_sum(hv);
            float tot = rl(g, 0) + rl(g, 8);         // uniform: row i's output
            if (lane == 0) out[blockIdx.x * 8 + i] = tot + fb2s;
        }
    }
}

extern "C" void kernel_launch(void* const* d_in, const int* in_sizes, int n_in,
                              void* d_out, int out_size, void* d_ws, size_t ws_size,
                              hipStream_t stream) {
    const float* x    = (const float*)d_in[0];
    const float* mW1  = (const float*)d_in[1];
    const float* mb1  = (const float*)d_in[2];
    const float* mW2  = (const float*)d_in[3];
    const float* mb2  = (const float*)d_in[4];
    const float* fW1  = (const float*)d_in[5];
    const float* fb1  = (const float*)d_in[6];
    const float* fW2  = (const float*)d_in[7];
    const float* fb2  = (const float*)d_in[8];
    float* out = (float*)d_out;

    const int rows = in_sizes[0] / 14;   // 64*1024 = 65536
    const int T    = 1024;               // x is (B,14,T)

    dim3 block(256);                     // 8 wave32 per block, 1 row per wave
    dim3 grid(rows / 8);
    maron_inv_kernel<<<grid, block, 0, stream>>>(x, mW1, mb1, mW2, mb2,
                                                 fW1, fb1, fW2, fb2,
                                                 out, rows, T);
}
